// GIN_2585570312520
// MI455X (gfx1250) — compile-verified, matrix operand order checked by reading, not run on hardware
//
#include <hip/hip_runtime.h>
#include <hip/hip_bf16.h>

// ---------------------------------------------------------------------------
// Types for CDNA5 WMMA
// ---------------------------------------------------------------------------
typedef __attribute__((ext_vector_type(16))) __bf16 v16bf;
typedef __attribute__((ext_vector_type(8)))  float  v8f;

union FragU {
    v16bf v;
    uint4 q[2];
};

__device__ __forceinline__ unsigned short f2bf_bits(float f) {
    union { float f; unsigned int u; } x;
    x.f = f;
    unsigned int u = x.u;
    // round-to-nearest-even bf16 truncation
    unsigned int r = u + 0x7FFFu + ((u >> 16) & 1u);
    return (unsigned short)(r >> 16);
}

// ---------------------------------------------------------------------------
// Weight convert+transpose: w [K][N] f32  ->  wt [N][K] bf16
// ---------------------------------------------------------------------------
__global__ void convert_wT(const float* __restrict__ w,
                           unsigned short* __restrict__ wt,
                           int K, int N) {
    int idx = blockIdx.x * blockDim.x + threadIdx.x;
    if (idx < N * K) {
        int ncol = idx / K;
        int k    = idx % K;
        wt[idx] = f2bf_bits(w[(size_t)k * N + ncol]);
    }
}

// ---------------------------------------------------------------------------
// Edge aggregation: agg[dst] += h[src]   (agg pre-initialized to h -> GIN x+sum)
// 32 lanes per edge, 4 floats per lane (coalesced float4 gather).
// ---------------------------------------------------------------------------
__global__ void agg_edges(float* __restrict__ agg,
                          const float* __restrict__ h,
                          const int* __restrict__ src,
                          const int* __restrict__ dst,
                          int E) {
    int tid = blockIdx.x * blockDim.x + threadIdx.x;
    int e = tid >> 5;          // edge index
    int g = tid & 31;          // 4-float group within the 128-d feature
    if (e < E) {
        int s = src[e];
        int d = dst[e];
        const float4 v = *(const float4*)(h + (size_t)s * 128 + g * 4);
        float* p = agg + (size_t)d * 128 + g * 4;
        atomicAdd(p + 0, v.x);
        atomicAdd(p + 1, v.y);
        atomicAdd(p + 2, v.z);
        atomicAdd(p + 3, v.w);
    }
}

// ---------------------------------------------------------------------------
// Fused GEMM + (bias | bias+BN) + optional ELU.
//   A   : [M][128] f32           (activations)
//   Bt  : [NT*16][128] bf16      (transposed weights; K = 128 fixed)
//   out : [M][NT*16] f32
// MODE: 0 = bias only, 1 = bias+ELU, 2 = bias+BN+ELU
// Block = 128 threads = 4 waves; each block does a 64 x N output tile.
// NT/MODE are compile-time so the v8f acc[NT] array stays fully in VGPRs.
// ---------------------------------------------------------------------------
template <int NT, int MODE>
__global__ __launch_bounds__(128)
void gemm_epilogue(const float* __restrict__ A,
                   const unsigned short* __restrict__ Bt,
                   const float* __restrict__ bias,
                   const float* __restrict__ bn_g,
                   const float* __restrict__ bn_b,
                   const float* __restrict__ bn_mu,
                   const float* __restrict__ bn_var,
                   float* __restrict__ out,
                   int M) {
    constexpr int N = NT * 16;
    __shared__ unsigned short ldsA[64 * 128];   // 16 KB
    __shared__ unsigned short ldsB[N * 128];    // <= 32 KB

    const int tid   = threadIdx.x;
    const int mBase = blockIdx.x * 64;

    // ---- stage Bt (N x 128 bf16) into LDS, 16B chunks ----
    {
        const uint4* gb = (const uint4*)(const void*)Bt;
        uint4* sb = (uint4*)(void*)ldsB;
        #pragma unroll
        for (int i = 0; i < N * 16 / 128; ++i) sb[i * 128 + tid] = gb[i * 128 + tid];
    }

    // ---- stage A tile (64 x 128 f32 -> bf16) into LDS ----
    #pragma unroll
    for (int it = 0; it < 16; ++it) {           // 64 rows * 32 float4 / 128 thr
        int i  = it * 128 + tid;
        int r  = i >> 5;
        int c4 = i & 31;
        int gr = mBase + r;
        float4 v;
        if (gr < M) v = *(const float4*)(A + (size_t)gr * 128 + c4 * 4);
        else        v = make_float4(0.f, 0.f, 0.f, 0.f);
        uint2 p;
        p.x = (unsigned)f2bf_bits(v.x) | ((unsigned)f2bf_bits(v.y) << 16);
        p.y = (unsigned)f2bf_bits(v.z) | ((unsigned)f2bf_bits(v.w) << 16);
        *(uint2*)(void*)&ldsA[r * 128 + c4 * 4] = p;
    }
    __syncthreads();

    const int wave    = tid >> 5;
    const int lane    = tid & 31;
    const int halfOff = (lane & 16) >> 1;           // +8 K-elements for upper half-wave
    const int rowA    = wave * 16 + (lane & 15);    // A-fragment row (M = 0..15 both halves)

    v8f acc[NT];
    #pragma unroll
    for (int n = 0; n < NT; ++n) acc[n] = v8f{};

    // ---- K loop: 4 steps of K=32, fully unrolled ----
    #pragma unroll
    for (int s = 0; s < 4; ++s) {
        FragU a;
        const unsigned short* ap = &ldsA[rowA * 128 + 32 * s + halfOff];
        a.q[0] = *(const uint4*)(const void*)(ap);
        a.q[1] = *(const uint4*)(const void*)(ap + 16);
        #pragma unroll
        for (int n = 0; n < NT; ++n) {
            FragU b;
            const unsigned short* bp =
                &ldsB[(n * 16 + (lane & 15)) * 128 + 32 * s + halfOff];
            b.q[0] = *(const uint4*)(const void*)(bp);
            b.q[1] = *(const uint4*)(const void*)(bp + 16);
            acc[n] = __builtin_amdgcn_wmma_f32_16x16x32_bf16(
                false, a.v, false, b.v, (short)0, acc[n], false, false);
        }
    }

    // ---- epilogue: bias / folded BN / ELU, C layout per ISA 7.12.2 ----
    const int rBase = mBase + wave * 16 + ((lane >> 4) << 3);
    #pragma unroll
    for (int n = 0; n < NT; ++n) {
        int col = n * 16 + (lane & 15);
        float bi = bias[col];
        float scale = 1.f, shift = 0.f;
        if (MODE == 2) {
            scale = bn_g[col] * rsqrtf(bn_var[col] + 1e-5f);
            shift = bn_b[col] - bn_mu[col] * scale;
        }
        #pragma unroll
        for (int r = 0; r < 8; ++r) {
            float v = acc[n][r] + bi;
            if (MODE == 2) v = v * scale + shift;
            if (MODE >= 1) v = (v > 0.f) ? v : (__expf(v) - 1.f);
            int gr = rBase + r;
            if (gr < M) out[(size_t)gr * N + col] = v;
        }
    }
}

// ---------------------------------------------------------------------------
// Host orchestration
// ---------------------------------------------------------------------------
extern "C" void kernel_launch(void* const* d_in, const int* in_sizes, int n_in,
                              void* d_out, int out_size, void* d_ws, size_t ws_size,
                              hipStream_t stream) {
    const float* x  = (const float*)d_in[0];
    const int*   ei = (const int*)d_in[1];
    const int Nn = in_sizes[0] / 128;
    const int E  = in_sizes[1] / 2;
    const int* src = ei;
    const int* dst = ei + E;

    // workspace carve
    char* w = (char*)d_ws;
    const size_t featBytes = (size_t)Nn * 128 * sizeof(float);
    float* bufA = (float*)w; w += featBytes;   // aggregation / final hidden
    float* bufB = (float*)w; w += featBytes;   // MLP hidden
    float* bufC = (float*)w; w += featBytes;   // layer output (ping)
    unsigned short* wT[8];
    const int wIn[8]   = {2, 8, 10, 16, 18, 24, 26, 28};  // c1w1,c1w2,c2w1,c2w2,c3w1,c3w2,lin1,lin2
    const int wNout[8] = {128, 128, 128, 128, 128, 128, 128, 64};
    for (int i = 0; i < 8; ++i) {
        wT[i] = (unsigned short*)w;
        w += (size_t)wNout[i] * 128 * sizeof(unsigned short);
    }

    // 1) weight convert+transpose (f32 -> bf16, N-major)
    for (int i = 0; i < 8; ++i) {
        int elems = 128 * wNout[i];
        convert_wT<<<(elems + 255) / 256, 256, 0, stream>>>(
            (const float*)d_in[wIn[i]], wT[i], 128, wNout[i]);
    }

    const int gemmGrid = (Nn + 63) / 64;
    const long long aggThreads = (long long)E * 32;
    const int aggGrid = (int)((aggThreads + 255) / 256);

    const float* cur = x;
    for (int l = 0; l < 3; ++l) {
        const int base = 2 + l * 8;
        const float* b1  = (const float*)d_in[base + 1];
        const float* g   = (const float*)d_in[base + 2];
        const float* bt  = (const float*)d_in[base + 3];
        const float* mu  = (const float*)d_in[base + 4];
        const float* var = (const float*)d_in[base + 5];
        const float* b2  = (const float*)d_in[base + 7];

        // agg = cur + segment_sum(cur[src] -> dst)
        hipMemcpyAsync(bufA, cur, featBytes, hipMemcpyDeviceToDevice, stream);
        agg_edges<<<aggGrid, 256, 0, stream>>>(bufA, cur, src, dst, E);

        // h = ELU(BN(agg @ w1 + b1))
        gemm_epilogue<8, 2><<<gemmGrid, 128, 0, stream>>>(
            bufA, wT[l * 2 + 0], b1, g, bt, mu, var, bufB, Nn);
        // h = ELU(h @ w2 + b2)
        gemm_epilogue<8, 1><<<gemmGrid, 128, 0, stream>>>(
            bufB, wT[l * 2 + 1], b2, b2, b2, b2, b2, bufC, Nn);
        cur = bufC;
    }

    // lin1: ELU(h @ w + b)
    const float* lb1 = (const float*)d_in[27];
    gemm_epilogue<8, 1><<<gemmGrid, 128, 0, stream>>>(
        bufC, wT[6], lb1, lb1, lb1, lb1, lb1, bufA, Nn);
    // lin2: h @ w + b  (N = 64, no activation)
    const float* lb2 = (const float*)d_in[29];
    gemm_epilogue<4, 0><<<gemmGrid, 128, 0, stream>>>(
        bufA, wT[7], lb2, lb2, lb2, lb2, lb2, (float*)d_out, Nn);
}
